// MultiHeadAttention_37185826849367
// MI455X (gfx1250) — compile-verified, matrix operand order checked by reading, not run on hardware
//
#include <hip/hip_runtime.h>
#include <hip/hip_bf16.h>

typedef __attribute__((ext_vector_type(16))) __bf16        v16bf;
typedef __attribute__((ext_vector_type(8)))  float         v8f;
typedef __attribute__((ext_vector_type(4)))  unsigned int  u32x4;
typedef __attribute__((ext_vector_type(8)))  int           i32x8;
typedef __attribute__((ext_vector_type(4)))  int           i32x4;

#define T_DIM 4096
#define C_DIM 128

__device__ __forceinline__ v8f wmma_bf16(v16bf a, v16bf b, v8f c) {
    // D = A(16x32 bf16) x B(32x16 bf16) + C(16x16 f32)
    return __builtin_amdgcn_wmma_f32_16x16x32_bf16(
        /*neg_a=*/false, a, /*neg_b=*/false, b,
        /*c_mod=*/(short)0, c, /*reuse_a=*/false, /*reuse_b=*/false);
}

// ---------------------------------------------------------------------------
// Tensor Data Mover: DMA one 32x32 f32 tile (row stride 4096 floats) from
// global into LDS, padding each 32-DWORD row with 4 DWORDs so LDS rows land
// on a 36-float pitch (bank-conflict-free strided reads, 16B aligned).
// D# layout per CDNA5 ISA; 6-arg builtin (clang-23).
// ---------------------------------------------------------------------------
__device__ __forceinline__ void tdm_load_tile_32x32(const float* gptr,
                                                    unsigned lds_byte_off) {
    const unsigned long long ga = (unsigned long long)(uintptr_t)gptr;
    u32x4 g0 = {
        1u,                                          // count=1, user descriptor
        lds_byte_off,                                // lds_addr
        (unsigned)(ga & 0xffffffffu),                // global_addr[31:0]
        (unsigned)((ga >> 32) & 0x01ffffffu)         // global_addr[56:32]
            | 0x80000000u                            // type = 2 ("image")
    };
    i32x8 g1 = {
        (int)(0x00020000u                            // data_size = 4B
              | (1u << 20)                           // pad_enable
              | (4u << 22)                           // pad_interval: 32 DWORDs
              | (3u << 25)),                         // pad_amount: 4 DWORDs
        (int)(4096u << 16),                          // tensor_dim0 = 4096 (lo16)
        (int)(32u << 16),                            // dim0 hi | tensor_dim1=32
        (int)(32u << 16),                            // dim1 hi | tile_dim0 = 32
        32,                                          // tile_dim1=32, tile_dim2=0
        4096,                                        // tensor_dim0_stride lo
        0, 0                                         // stride hi, dim1_stride
    };
    i32x4 z4 = {0, 0, 0, 0};                         // 2-D tile: groups 2/3 off
    i32x8 z8 = {0, 0, 0, 0, 0, 0, 0, 0};
    __builtin_amdgcn_tensor_load_to_lds(g0, g1, z4, z4, z8, 0);
}

// ---------------------------------------------------------------------------
// Kernel 1: GroupNorm. grid = B*32 groups, block = 256.
// ---------------------------------------------------------------------------
__global__ __launch_bounds__(256) void gn_kernel(
    const float* __restrict__ x, const float* __restrict__ gw,
    const float* __restrict__ gb, float* __restrict__ xn) {
    const int b = blockIdx.x >> 5;
    const int g = blockIdx.x & 31;
    const int CG = 4;
    const int N  = CG * T_DIM;
    const float* xp = x + ((size_t)b * C_DIM + g * CG) * T_DIM;

    float s = 0.f, ss = 0.f;
    for (int i = threadIdx.x; i < N; i += 256) {
        float v = xp[i];
        s += v; ss += v * v;
    }
    #pragma unroll
    for (int off = 16; off; off >>= 1) {
        s  += __shfl_xor(s,  off);
        ss += __shfl_xor(ss, off);
    }
    __shared__ float bs[8], bss[8];
    const int wid = threadIdx.x >> 5;
    if ((threadIdx.x & 31) == 0) { bs[wid] = s; bss[wid] = ss; }
    __syncthreads();
    s  = bs [threadIdx.x & 7];
    ss = bss[threadIdx.x & 7];
    #pragma unroll
    for (int off = 4; off; off >>= 1) {
        s  += __shfl_xor(s,  off);
        ss += __shfl_xor(ss, off);
    }
    const float inv_n = 1.0f / (float)N;
    const float mean  = s * inv_n;
    const float var   = ss * inv_n - mean * mean;
    const float rstd  = rsqrtf(var + 1e-5f);

    float* xnp = xn + ((size_t)b * C_DIM + g * CG) * T_DIM;
    for (int i = threadIdx.x; i < N; i += 256) {
        int c = g * CG + (i >> 12);
        xnp[i] = (xp[i] - mean) * rstd * gw[c] + gb[c];
    }
}

// ---------------------------------------------------------------------------
// Kernel 2/4: channel GEMM  D[b,o,t] = sum_c W[o,c]*X[b,c,t] + bias[o] (+res)
// One wave computes a 16x64 (o,t) strip: one A-operand per K-step feeds 4
// WMMAs (1x4 N-blocking). K stepped by 32 with bf16 WMMA, f32 accumulate.
// ---------------------------------------------------------------------------
template <bool RES>
__global__ __launch_bounds__(256) void gemm_cw_kernel(
    const float* __restrict__ W, const float* __restrict__ X,
    const float* __restrict__ bias, const float* __restrict__ res,
    float* __restrict__ D, int M, int K, int tilesM) {
    const int tilesN    = T_DIM / 64;                // 64 strips of 4 tiles
    const int tilesPerB = tilesM * tilesN;
    const int wave = blockIdx.x * (blockDim.x >> 5) + (threadIdx.x >> 5);
    const int b  = wave / tilesPerB;
    const int r  = wave % tilesPerB;
    const int mt = r / tilesN;
    const int nt = r % tilesN;

    const int lane = threadIdx.x & 31;
    const int half = lane >> 4;
    const int l16  = lane & 15;
    const int hk   = half * 8;

    const int o_row  = mt * 16 + l16;
    const int t_base = nt * 64 + l16;
    const float* wrow = W + (size_t)o_row * K;
    const float* xcol = X + (size_t)b * K * T_DIM + t_base;

    v8f acc[4] = {{}, {}, {}, {}};
    for (int kb = 0; kb < K; kb += 32) {
        v16bf a;
        #pragma unroll
        for (int e = 0; e < 16; ++e) {
            const int kk = kb + ((e < 8) ? 0 : 16) + hk + (e & 7);
            a[e] = (__bf16)wrow[kk];
        }
        #pragma unroll
        for (int sub = 0; sub < 4; ++sub) {
            v16bf bm;
            #pragma unroll
            for (int e = 0; e < 16; ++e) {
                const int kk = kb + ((e < 8) ? 0 : 16) + hk + (e & 7);
                bm[e] = (__bf16)xcol[(size_t)kk * T_DIM + sub * 16];
            }
            acc[sub] = wmma_bf16(a, bm, acc[sub]);
        }
    }

    #pragma unroll
    for (int sub = 0; sub < 4; ++sub) {
        #pragma unroll
        for (int rI = 0; rI < 8; ++rI) {
            const int o = mt * 16 + rI + hk;
            float v = acc[sub][rI] + bias[o];
            const size_t idx = ((size_t)b * M + o) * T_DIM + t_base + sub * 16;
            if (RES) v += res[idx];
            D[idx] = v;
        }
    }
}

// ---------------------------------------------------------------------------
// Kernel 3: flash attention over qkv [B, 384, T].
// Per (b,h): q = rows [h*96, +32), k = +32, v = +64 (reference reshape order).
// grid = 16 bh * 32 qblocks, block = 256 (8 waves x 16 queries).
// K/V 32x32 tiles staged to LDS by the Tensor Data Mover (wave 0 issues,
// s_wait_tensorcnt, barrier). Per key-block:
//   S^T = WMMA(A=K[s,d], B=Q[d,t])  -> softmax axis on C's M dim (lane-local)
//   O^T += WMMA(A=V[d,s], B=P^T)    -> P^T packs in-lane from S^T C-layout
// ---------------------------------------------------------------------------
__global__ __launch_bounds__(256) void attn_kernel(
    const float* __restrict__ qkv, float* __restrict__ attnO) {
    const int bh   = blockIdx.x >> 5;
    const int qblk = blockIdx.x & 31;
    const int b = bh >> 2, h = bh & 3;

    const float* qp = qkv + ((size_t)b * 384 + h * 96) * T_DIM;
    const float* kp = qp + 32 * (size_t)T_DIM;
    const float* vp = kp + 32 * (size_t)T_DIM;

    const int wid  = threadIdx.x >> 5;
    const int lane = threadIdx.x & 31;
    const int half = lane >> 4;
    const int l16  = lane & 15;
    const int hk   = half * 8;
    const int t0   = qblk * 128 + wid * 16;
    const float scale = 0.42044820762685725f;        // 32^(-1/4)

    // 32 rows x 36-float pitch (TDM pads 4 DWORDs per 32-DWORD row)
    __shared__ float ksh[32 * 36];
    __shared__ float vsh[32 * 36];
    const unsigned koff = (unsigned)(uintptr_t)(void*)ksh;
    const unsigned voff = (unsigned)(uintptr_t)(void*)vsh;

    // Q tile as WMMA B operand (n = t, k = d), loaded once.
    v16bf bq;
    {
        const float* qc = qp + (t0 + l16);
        #pragma unroll
        for (int e = 0; e < 16; ++e) {
            const int d = ((e < 8) ? 0 : 16) + hk + (e & 7);
            bq[e] = (__bf16)(qc[(size_t)d * T_DIM] * scale);
        }
    }

    v8f  o0 = {}, o1 = {};
    float m_t = -3.0e38f, l_t = 0.f;

    for (int s0 = 0; s0 < T_DIM; s0 += 32) {
        if (wid == 0) {                              // one wave drives the TDM
            tdm_load_tile_32x32(kp + s0, koff);
            tdm_load_tile_32x32(vp + s0, voff);
            __builtin_amdgcn_s_wait_tensorcnt(0);
        }
        __syncthreads();                             // publish tiles to all waves

        v16bf ak0, ak1, av0, av1;
        #pragma unroll
        for (int e = 0; e < 16; ++e) {
            const int kk = ((e < 8) ? 0 : 16) + hk + (e & 7);
            ak0[e] = (__bf16)(ksh[kk * 36 + l16]      * scale); // K: m=s,   k=d
            ak1[e] = (__bf16)(ksh[kk * 36 + 16 + l16] * scale); // K: m=s+16
            av0[e] = (__bf16)vsh[l16 * 36 + kk];                // V: m=d,   k=s
            av1[e] = (__bf16)vsh[(16 + l16) * 36 + kk];         // V: m=d+16
        }
        __syncthreads();                             // tiles consumed

        v8f st0 = {}, st1 = {};
        st0 = wmma_bf16(ak0, bq, st0);               // S^T rows s=0..15
        st1 = wmma_bf16(ak1, bq, st1);               // S^T rows s=16..31

        float bm = st0[0];
        #pragma unroll
        for (int rI = 1; rI < 8; ++rI) bm = fmaxf(bm, st0[rI]);
        #pragma unroll
        for (int rI = 0; rI < 8; ++rI) bm = fmaxf(bm, st1[rI]);
        bm = fmaxf(bm, __shfl_xor(bm, 16));
        const float m_new = fmaxf(m_t, bm);
        const float alpha = __expf(m_t - m_new);

        float psum = 0.f;
        v8f p0, p1;
        #pragma unroll
        for (int rI = 0; rI < 8; ++rI) {
            p0[rI] = __expf(st0[rI] - m_new);
            p1[rI] = __expf(st1[rI] - m_new);
            psum += p0[rI] + p1[rI];
        }
        psum += __shfl_xor(psum, 16);
        l_t = l_t * alpha + psum;
        m_t = m_new;
        #pragma unroll
        for (int rI = 0; rI < 8; ++rI) { o0[rI] *= alpha; o1[rI] *= alpha; }

        v16bf pb;                                    // P^T -> B-layout, in-lane
        #pragma unroll
        for (int e = 0; e < 16; ++e)
            pb[e] = (e < 8) ? (__bf16)p0[e] : (__bf16)p1[e - 8];

        o0 = wmma_bf16(av0, pb, o0);
        o1 = wmma_bf16(av1, pb, o1);
    }

    const float inv_l = 1.0f / l_t;
    float* op = attnO + ((size_t)b * C_DIM + h * 32) * T_DIM + (t0 + l16);
    #pragma unroll
    for (int rI = 0; rI < 8; ++rI) {
        op[(size_t)(rI + hk) * T_DIM]      = o0[rI] * inv_l;
        op[(size_t)(16 + rI + hk) * T_DIM] = o1[rI] * inv_l;
    }
}

// ---------------------------------------------------------------------------
// Host-side launcher
// ---------------------------------------------------------------------------
extern "C" void kernel_launch(void* const* d_in, const int* in_sizes, int n_in,
                              void* d_out, int out_size, void* d_ws, size_t ws_size,
                              hipStream_t stream) {
    const float* x      = (const float*)d_in[0];
    const float* gn_w   = (const float*)d_in[1];
    const float* gn_b   = (const float*)d_in[2];
    const float* qkv_w  = (const float*)d_in[3];
    const float* qkv_b  = (const float*)d_in[4];
    const float* proj_w = (const float*)d_in[5];
    const float* proj_b = (const float*)d_in[6];
    float* out = (float*)d_out;

    // workspace: xn [4,128,4096] | qkv [4,384,4096] | attn [4,128,4096] = 40 MB
    float* xn   = (float*)d_ws;
    float* qkvb = xn   + (size_t)4 * 128 * T_DIM;
    float* attn = qkvb + (size_t)4 * 384 * T_DIM;

    // 1) GroupNorm
    gn_kernel<<<128, 256, 0, stream>>>(x, gn_w, gn_b, xn);

    // 2) QKV GEMM: 4 b x 24 Mtiles x 64 Nstrips = 6144 waves = 768 blocks
    gemm_cw_kernel<false><<<768, 256, 0, stream>>>(qkv_w, xn, qkv_b, nullptr,
                                                   qkvb, 384, 128, 24);

    // 3) Flash attention: 16 bh x 32 query-blocks
    attn_kernel<<<512, 256, 0, stream>>>(qkvb, attn);

    // 4) Proj GEMM + bias + residual: 4 b x 8 x 64 = 2048 waves = 256 blocks
    gemm_cw_kernel<true><<<256, 256, 0, stream>>>(proj_w, attn, proj_b, x,
                                                  out, 128, 128, 8);
}
